// MultiHeadedAttention_82489141887732
// MI455X (gfx1250) — compile-verified, matrix operand order checked by reading, not run on hardware
//
#include <hip/hip_runtime.h>

// ---------------------------------------------------------------------------
// MultiHeadedAttention for MI455X (gfx1250, wave32, WMMA bf16 16x16x32)
// B=4, S=1024, D=1024, H=16, DK=64
//   1) convert f32 -> bf16 (q,k,v inputs, Wq,Wk,Wv,Wo)
//   2) proj GEMMs (WMMA bf16): Q,K -> [B,H,S,DK], V -> [B,H,DK,S] (transposed)
//   3) flash attention, one wave per (b,h,16-row q-tile); softmax done in a
//      row-per-lane transposed domain via LDS (no cross-lane shuffles);
//      K and V fragments preloaded to overlap global latency with WMMA/softmax
//   4) output GEMM (WMMA bf16, f32 out): X @ Wo^T + bo
// ---------------------------------------------------------------------------

typedef __attribute__((ext_vector_type(16))) __bf16 v16bf;
typedef __attribute__((ext_vector_type(8)))  __bf16 v8bf;
typedef __attribute__((ext_vector_type(4)))  __bf16 v4bf;
typedef __attribute__((ext_vector_type(8)))  float  v8f;
typedef __attribute__((ext_vector_type(4)))  float  v4f;

#define S_LEN  1024
#define DMODEL 1024
#define NHEAD  16
#define DKH    64
#define BATCH  4

static __device__ __forceinline__ v8bf ld8(const __bf16* p) {
    return *reinterpret_cast<const v8bf*>(p);
}
static __device__ __forceinline__ v16bf cat8(v8bf lo, v8bf hi) {
    v16bf r;
#pragma unroll
    for (int i = 0; i < 8; ++i) { r[i] = lo[i]; r[i + 8] = hi[i]; }
    return r;
}

// --------------------------- f32 -> bf16 convert ---------------------------
__global__ void cvt_bf16_kernel(const float* __restrict__ in,
                                __bf16* __restrict__ out, int n) {
    int i = (blockIdx.x * blockDim.x + threadIdx.x) * 4;
    if (i >= n) return;
    float4 v = *reinterpret_cast<const float4*>(in + i);
    v4bf o;
    o[0] = (__bf16)v.x; o[1] = (__bf16)v.y; o[2] = (__bf16)v.z; o[3] = (__bf16)v.w;
    *reinterpret_cast<v4bf*>(out + i) = o;
}

// ----------------------------- projection GEMM -----------------------------
// C[m,n] = sum_k X[m,k] * W[n,k] + bias[n]
// MODE 0: bf16 -> [B,H,S,DK]   (q,k)
// MODE 1: bf16 -> [B,H,DK,S]   (v transposed, for P*V B-operand)
// MODE 2: f32  -> [B*S, D]     (final output)
template <int MODE>
__global__ __launch_bounds__(256, 1)
void proj_kernel(const __bf16* __restrict__ X,
                 const __bf16* __restrict__ W,
                 const float*  __restrict__ bias,
                 void* __restrict__ outp) {
    const int gw   = (blockIdx.x * blockDim.x + threadIdx.x) >> 5;
    const int lane = threadIdx.x & 31;
    const int r    = lane & 15;
    const int half = lane >> 4;
    const int m0 = (gw >> 4) * 16;   // 256 m-tiles
    const int n0 = (gw & 15) * 64;   // 16 n-groups

    v8f acc[4] = {};
    const __bf16* xrow = X + (size_t)(m0 + r) * DMODEL;

    for (int k0 = 0; k0 < DMODEL; k0 += 32) {
        v16bf a = cat8(ld8(xrow + k0 + half * 8),
                       ld8(xrow + k0 + 16 + half * 8));
#pragma unroll
        for (int nt = 0; nt < 4; ++nt) {
            const __bf16* wrow = W + (size_t)(n0 + nt * 16 + r) * DMODEL;
            v16bf b = cat8(ld8(wrow + k0 + half * 8),
                           ld8(wrow + k0 + 16 + half * 8));
            acc[nt] = __builtin_amdgcn_wmma_f32_16x16x32_bf16(
                false, a, false, b, (short)0, acc[nt], false, false);
        }
    }

    // C layout: lanes 0-15 -> M=j, lanes 16-31 -> M=j+8; N = lane&15
#pragma unroll
    for (int nt = 0; nt < 4; ++nt) {
        const int nfeat = n0 + nt * 16 + r;
        const float bv = bias[nfeat];
#pragma unroll
        for (int j = 0; j < 8; ++j) {
            const int m = m0 + j + half * 8;
            const float val = acc[nt][j] + bv;
            if (MODE == 2) {
                ((float*)outp)[(size_t)m * DMODEL + nfeat] = val;
            } else {
                const int h  = nfeat >> 6;
                const int dk = nfeat & 63;
                const int bb = m >> 10;
                const int s  = m & 1023;
                __bf16* ob = (__bf16*)outp;
                if (MODE == 0)
                    ob[(((size_t)(bb * NHEAD + h) * S_LEN) + s) * DKH + dk] = (__bf16)val;
                else
                    ob[(((size_t)(bb * NHEAD + h) * DKH) + dk) * S_LEN + s] = (__bf16)val;
            }
        }
    }
}

// ------------------------------ flash attention ----------------------------
#define S_STRIDE 36   // f32 score-tile row stride (bank-spread, 16B aligned)
#define P_STRIDE 40   // bf16 P-tile row stride   (bank-spread, 16B aligned)

struct AttnLds {
    float  s[16 * S_STRIDE];   // raw score tile (f32)
    __bf16 p[16 * P_STRIDE];   // p*g tile (bf16, A-operand staging)
    float  alpha[16];          // per-row online-softmax rescale
    float  linv[16];           // per-row 1/denominator
};

__global__ __launch_bounds__(256, 1)
void attn_kernel(const __bf16* __restrict__ Q,
                 const __bf16* __restrict__ K,
                 const __bf16* __restrict__ Vt,
                 const int*    __restrict__ mask,
                 const float*  __restrict__ gp,
                 __bf16* __restrict__ Xa) {
    __shared__ __align__(64) AttnLds lds[8];
    AttnLds& L = lds[threadIdx.x >> 5];

    const int gw   = (blockIdx.x * blockDim.x + threadIdx.x) >> 5;
    const int lane = threadIdx.x & 31;
    const int r    = lane & 15;
    const int half = lane >> 4;

    const int bh = gw >> 6;          // 64 q-tiles per (b,h)
    const int qt = gw & 63;
    const int b  = bh >> 4;
    const int h  = bh & 15;
    const int m0 = qt * 16;

    const __bf16* Qbh = Q  + (size_t)bh * S_LEN * DKH;
    const __bf16* Kbh = K  + (size_t)bh * S_LEN * DKH;
    const __bf16* Vbh = Vt + (size_t)bh * DKH * S_LEN;

    // Q tile 16x64 as two A fragments (K = 0..31, 32..63)
    const __bf16* qrow = Qbh + (size_t)(m0 + r) * DKH;
    const v16bf aq0 = cat8(ld8(qrow + half * 8),      ld8(qrow + 16 + half * 8));
    const v16bf aq1 = cat8(ld8(qrow + 32 + half * 8), ld8(qrow + 48 + half * 8));

    v8f   o[4] = {};
    float mold = -1e30f, lsum = 0.0f;        // valid in lanes 0..15 (row = r)
    const int qrow_idx = m0 + r;             // softmax row owned by this lane
    const int* __restrict__ mbase = mask + ((size_t)b * S_LEN + qrow_idx) * S_LEN;
    const float* __restrict__ gbase = gp + ((size_t)b * S_LEN + qrow_idx) * S_LEN;

    for (int kt = 0; kt < S_LEN / 32; ++kt) {
        const int k0 = kt * 32;

        // ---- (a) preload ALL K fragments (distinct regs -> one big clause)
        const __bf16* krow0 = Kbh + (size_t)(k0 + r) * DKH;
        const __bf16* krow1 = Kbh + (size_t)(k0 + 16 + r) * DKH;
        const v16bf bk00 = cat8(ld8(krow0 + half * 8),      ld8(krow0 + 16 + half * 8));
        const v16bf bk01 = cat8(ld8(krow0 + 32 + half * 8), ld8(krow0 + 48 + half * 8));
        const v16bf bk10 = cat8(ld8(krow1 + half * 8),      ld8(krow1 + 16 + half * 8));
        const v16bf bk11 = cat8(ld8(krow1 + 32 + half * 8), ld8(krow1 + 48 + half * 8));

        // ---- (b) preload V fragments early: latency hidden under softmax --
        v16bf bv[4];
#pragma unroll
        for (int nt = 0; nt < 4; ++nt) {
            const __bf16* vrow = Vbh + (size_t)(nt * 16 + r) * S_LEN + k0;
            bv[nt] = cat8(ld8(vrow + half * 8), ld8(vrow + 16 + half * 8));
        }

        // ---- (c) scores 16x32 = two 16x16 WMMA n-tiles --------------------
        v8f st[2];
        {
            v8f s0 = {}, s1 = {};
            s0 = __builtin_amdgcn_wmma_f32_16x16x32_bf16(false, aq0, false, bk00,
                                                         (short)0, s0, false, false);
            s1 = __builtin_amdgcn_wmma_f32_16x16x32_bf16(false, aq0, false, bk10,
                                                         (short)0, s1, false, false);
            s0 = __builtin_amdgcn_wmma_f32_16x16x32_bf16(false, aq1, false, bk01,
                                                         (short)0, s0, false, false);
            s1 = __builtin_amdgcn_wmma_f32_16x16x32_bf16(false, aq1, false, bk11,
                                                         (short)0, s1, false, false);
            st[0] = s0; st[1] = s1;
        }

        // ---- (d) dump raw score tile to LDS (C layout -> row major) -------
#pragma unroll
        for (int nt = 0; nt < 2; ++nt)
#pragma unroll
            for (int j = 0; j < 8; ++j)
                L.s[(j + half * 8) * S_STRIDE + nt * 16 + r] = st[nt][j];

        asm volatile("s_wait_dscnt 0" ::: "memory");

        // ---- (e) row-per-lane softmax (lanes 0..15, zero shuffles) --------
        if (half == 0) {
            const float* srow = &L.s[r * S_STRIDE];
            const int4*   mrow = (const int4*)(mbase + k0);
            const float4* grow = (const float4*)(gbase + k0);

            // pass 1: mask + scale + row max (mask loaded at point of use)
            float sc[32];
            float tmax = -1e30f;
#pragma unroll
            for (int c4 = 0; c4 < 8; ++c4) {
                const v4f  sv = *(const v4f*)(srow + c4 * 4);
                const int4 m4 = mrow[c4];
                const int  cb = k0 + c4 * 4;
#pragma unroll
                for (int e = 0; e < 4; ++e) {
                    const int  mv = (e == 0) ? m4.x : (e == 1) ? m4.y
                                  : (e == 2) ? m4.z : m4.w;
                    const bool al = (mv != 0) || (qrow_idx == cb + e);
                    const float v = al ? sv[e] * 0.125f : -1e9f;
                    sc[c4 * 4 + e] = v;
                    tmax = fmaxf(tmax, v);
                }
            }
            const float mnew = fmaxf(mold, tmax);
            const float alp  = __expf(mold - mnew);
            mold = mnew;

            // pass 2: exp, row sum, group_prob weighting (gp at point of use)
            float sum = 0.0f;
            __bf16* prow = &L.p[r * P_STRIDE];
#pragma unroll
            for (int c8 = 0; c8 < 4; ++c8) {
                const float4 ga = grow[c8 * 2];
                const float4 gb = grow[c8 * 2 + 1];
                v8bf pv;
#pragma unroll
                for (int e = 0; e < 8; ++e) {
                    const float p = __expf(sc[c8 * 8 + e] - mnew);
                    sum += p;
                    const float g = (e == 0) ? ga.x : (e == 1) ? ga.y
                                  : (e == 2) ? ga.z : (e == 3) ? ga.w
                                  : (e == 4) ? gb.x : (e == 5) ? gb.y
                                  : (e == 6) ? gb.z : gb.w;
                    pv[e] = (__bf16)(p * g);
                }
                *(v8bf*)(prow + c8 * 8) = pv;
            }
            lsum = lsum * alp + sum;
            L.alpha[r] = alp;
        }

        asm volatile("s_wait_dscnt 0" ::: "memory");

        // ---- (f) rescale accumulators, fetch P as A-fragment --------------
        const v4f a0 = *(const v4f*)(&L.alpha[half * 8]);
        const v4f a1 = *(const v4f*)(&L.alpha[half * 8 + 4]);
#pragma unroll
        for (int nt = 0; nt < 4; ++nt)
#pragma unroll
            for (int j = 0; j < 8; ++j)
                o[nt][j] *= (j < 4) ? a0[j] : a1[j - 4];

        const __bf16* prow = &L.p[r * P_STRIDE];
        const v16bf ap = cat8(ld8(prow + half * 8), ld8(prow + 16 + half * 8));

        // ---- (g) o(16x64) += P(16x32) x V(32x64), V preloaded -------------
#pragma unroll
        for (int nt = 0; nt < 4; ++nt)
            o[nt] = __builtin_amdgcn_wmma_f32_16x16x32_bf16(
                false, ap, false, bv[nt], (short)0, o[nt], false, false);
    }

    // ---- finalize: 1/denominator via LDS bounce, write [B,S,D] bf16 -------
    if (half == 0) L.linv[r] = 1.0f / lsum;
    asm volatile("s_wait_dscnt 0" ::: "memory");
    const v4f l0 = *(const v4f*)(&L.linv[half * 8]);
    const v4f l1 = *(const v4f*)(&L.linv[half * 8 + 4]);
#pragma unroll
    for (int j = 0; j < 8; ++j) {
        const int   row = m0 + j + half * 8;
        const float inv = (j < 4) ? l0[j] : l1[j - 4];
#pragma unroll
        for (int nt = 0; nt < 4; ++nt) {
            const int d = h * DKH + nt * 16 + r;
            Xa[((size_t)b * S_LEN + row) * DMODEL + d] = (__bf16)(o[nt][j] * inv);
        }
    }
}

// ------------------------------- launcher ----------------------------------
extern "C" void kernel_launch(void* const* d_in, const int* in_sizes, int n_in,
                              void* d_out, int out_size, void* d_ws, size_t ws_size,
                              hipStream_t stream) {
    const float* query = (const float*)d_in[0];
    const float* key   = (const float*)d_in[1];
    const float* value = (const float*)d_in[2];
    const int*   mask  = (const int*)d_in[3];
    const float* gp    = (const float*)d_in[4];
    const float* Wq    = (const float*)d_in[5];
    const float* bq    = (const float*)d_in[6];
    const float* Wk    = (const float*)d_in[7];
    const float* bk    = (const float*)d_in[8];
    const float* Wv    = (const float*)d_in[9];
    const float* bv    = (const float*)d_in[10];
    const float* Wo    = (const float*)d_in[11];
    const float* bo    = (const float*)d_in[12];

    char* ws = (char*)d_ws;
    const size_t NX = (size_t)BATCH * S_LEN * DMODEL;   // 4,194,304
    const size_t NW = (size_t)DMODEL * DMODEL;          // 1,048,576
    __bf16* qb  = (__bf16*)(ws);
    __bf16* kb  = (__bf16*)(ws + 2 * NX);
    __bf16* vb  = (__bf16*)(ws + 4 * NX);
    __bf16* wqb = (__bf16*)(ws + 6 * NX);
    __bf16* wkb = (__bf16*)(ws + 6 * NX + 2 * NW);
    __bf16* wvb = (__bf16*)(ws + 6 * NX + 4 * NW);
    __bf16* wob = (__bf16*)(ws + 6 * NX + 6 * NW);
    __bf16* Qh  = (__bf16*)(ws + 6 * NX + 8 * NW);
    __bf16* Kh  = (__bf16*)(ws + 8 * NX + 8 * NW);
    __bf16* Vt  = (__bf16*)(ws + 10 * NX + 8 * NW);
    __bf16* Xa  = (__bf16*)(ws + 12 * NX + 8 * NW);

    // 1) bf16 conversions
    cvt_bf16_kernel<<<(int)(NX / 4 / 256), 256, 0, stream>>>(query, qb, (int)NX);
    cvt_bf16_kernel<<<(int)(NX / 4 / 256), 256, 0, stream>>>(key,   kb, (int)NX);
    cvt_bf16_kernel<<<(int)(NX / 4 / 256), 256, 0, stream>>>(value, vb, (int)NX);
    cvt_bf16_kernel<<<(int)(NW / 4 / 256), 256, 0, stream>>>(Wq, wqb, (int)NW);
    cvt_bf16_kernel<<<(int)(NW / 4 / 256), 256, 0, stream>>>(Wk, wkb, (int)NW);
    cvt_bf16_kernel<<<(int)(NW / 4 / 256), 256, 0, stream>>>(Wv, wvb, (int)NW);
    cvt_bf16_kernel<<<(int)(NW / 4 / 256), 256, 0, stream>>>(Wo, wob, (int)NW);

    // 2) projections: 4096 waves = 512 blocks x 8 waves
    proj_kernel<0><<<512, 256, 0, stream>>>(qb, wqb, bq, (void*)Qh);
    proj_kernel<0><<<512, 256, 0, stream>>>(kb, wkb, bk, (void*)Kh);
    proj_kernel<1><<<512, 256, 0, stream>>>(vb, wvb, bv, (void*)Vt);

    // 3) attention: one wave per (b,h,q-tile) -> 4096 waves
    attn_kernel<<<512, 256, 0, stream>>>(Qh, Kh, Vt, mask, gp, Xa);

    // 4) output projection (f32 result)
    proj_kernel<2><<<512, 256, 0, stream>>>(Xa, wob, bo, d_out);
}
